// MambaFormerBlock_40664750358847
// MI455X (gfx1250) — matrix operand order repaired and checked by reading, not compiled, not run on hardware
//
#include <hip/hip_runtime.h>
#include <cstdint>
#include <cstddef>

#define DIMC   64
#define DINNER 128
#define DSTATE 16
#define HIDN   170
#define BATCH  2
#define HW     128
#define LSEQ   (HW*HW)        /* 16384 */
#define MTOK   (BATCH*LSEQ)   /* 32768 */
#define EPSV   1e-5f

typedef unsigned short ushort_t;
typedef __attribute__((ext_vector_type(16))) __bf16 v16bf;
typedef __attribute__((ext_vector_type(8)))  float  v8f;
typedef __attribute__((ext_vector_type(4)))  unsigned int u32x4;
typedef __attribute__((ext_vector_type(8)))  int i32x8;
typedef __attribute__((ext_vector_type(4)))  int i32x4;

#if __has_include(<hip/amd_detail/amd_gfx1250_TDM.h>)
#define TDM_6ARG 1
#endif

__device__ __forceinline__ ushort_t f2bf(float f) {
  unsigned x = __float_as_uint(f);
  unsigned r = (x + 0x7FFFu + ((x >> 16) & 1u)) >> 16;
  return (ushort_t)r;
}
__device__ __forceinline__ float sigm(float x) { return 1.0f / (1.0f + __expf(-x)); }

// ---------------------------------------------------------------------------
// Tensor Data Mover: load a [rows x K] bf16 tile (row-major, stride K) from
// global into LDS at lds_off.  D# packing per CDNA5 ISA ch.8:
//  group0: count=1 | lds_addr | global_addr[56:0] | type=2
//  group1: data_size=1(2B) | tensor_dim0=K | tensor_dim1=rows |
//          tile_dim0=K | tile_dim1=rows | tensor_dim0_stride=K
// ---------------------------------------------------------------------------
__device__ __forceinline__ void tdm_load_tile_bf16(const ushort_t* gptr, unsigned lds_off,
                                                   int K, int rows) {
  unsigned long long ga = (unsigned long long)(uintptr_t)gptr;
  u32x4 g0;
  g0[0] = 1u;                                              // count=1 (user D#)
  g0[1] = lds_off;                                         // lds_addr (bytes)
  g0[2] = (unsigned)(ga & 0xFFFFFFFFu);                    // global_addr[31:0]
  g0[3] = (unsigned)((ga >> 32) & 0x01FFFFFFu) | (2u << 30); // ga[56:32] | type=2
  i32x8 g1;
  unsigned uk = (unsigned)K, ur = (unsigned)rows;
  g1[0] = (int)(1u << 16);                                 // data_size=1 -> 2 bytes
  g1[1] = (int)((uk & 0xFFFFu) << 16);                     // tensor_dim0 lo at [63:48]
  g1[2] = (int)((uk >> 16) | ((ur & 0xFFFFu) << 16));      // tdim0 hi | tensor_dim1 lo
  g1[3] = (int)((ur >> 16) | ((uk & 0xFFFFu) << 16));      // tdim1 hi | tile_dim0
  g1[4] = (int)(ur & 0xFFFFu);                             // tile_dim1 | tile_dim2=0
  g1[5] = (int)uk;                                         // tensor_dim0_stride[31:0]
  g1[6] = 0;                                               // stride0 hi | stride1 lo
  g1[7] = 0;
  i32x4 z4 = {0, 0, 0, 0};
#if defined(TDM_6ARG)
  i32x8 z8 = {0, 0, 0, 0, 0, 0, 0, 0};
  __builtin_amdgcn_tensor_load_to_lds(g0, g1, z4, z4, z8, 0);
#else
  __builtin_amdgcn_tensor_load_to_lds(g0, g1, z4, z4, 0);
#endif
}

// ---------------------------------------------------------------------------
// Weight convert + zero-pad to bf16:  dst[n*Kdst+k] = (n<Nsrc && k<Ksrc)?src:0
// ---------------------------------------------------------------------------
__global__ void cvt_pad_kernel(const float* __restrict__ src, ushort_t* __restrict__ dst,
                               int Nsrc, int Ksrc, int Kdst, int total) {
  int gid = blockIdx.x * blockDim.x + threadIdx.x;
  if (gid >= total) return;
  int n = gid / Kdst, k = gid % Kdst;
  dst[gid] = (n < Nsrc && k < Ksrc) ? f2bf(src[(size_t)n * Ksrc + k]) : (ushort_t)0;
}

// ---------------------------------------------------------------------------
// WMMA GEMM:  C[M x Npad] = A[M x K](bf16) * W[Npad x K]^T(bf16), f32 accum.
// Block = 256 thr = 8 waves; block tile 128(M) x 64(N).
// Wave w owns M rows [16w,16w+16) and all 4 N sub-tiles (4 accumulators,
// 4 WMMAs per k-step reusing one A fragment).
// Whole 128 x K A tile is DMA'd to LDS once by the Tensor Data Mover.
// planar!=0 stores C[n*ldc + m] (channel-planar) instead of C[m*ldc + n].
// ---------------------------------------------------------------------------
__global__ __launch_bounds__(256)
void wmma_gemm_bf16(const ushort_t* __restrict__ A, const ushort_t* __restrict__ W,
                    float* __restrict__ C, int K, int ldc, int planar) {
  __shared__ ushort_t lA[128 * 192];     // max K = 192 -> 48 KB (WGP has 320 KB)
  const int lane   = threadIdx.x & 31;
  const int wave   = threadIdx.x >> 5;
  const int blockM = blockIdx.x * 128;
  const int nbase  = blockIdx.y * 64;
  const int half   = lane >> 4;
  const int l16    = lane & 15;

  // One TDM op loads the entire A block tile; wave0 issues, all waves sync.
  if (threadIdx.x < 32) {
    tdm_load_tile_bf16(A + (size_t)blockM * K, (unsigned)(uintptr_t)&lA[0], K, 128);
    __builtin_amdgcn_s_wait_tensorcnt(0);
  }
  __builtin_prefetch(W + (size_t)(nbase + l16) * K, 0, 1);
  __syncthreads();

  v8f acc[4] = {};
  for (int kk = 0; kk < K; kk += 32) {
    union { v16bf v; unsigned u[8]; } af;
#pragma unroll
    for (int v = 0; v < 8; ++v) {
      // A 16-bit 16x32 fragment layout (ISA 7.12.2)
      int ka = (v >> 2) * 16 + half * 8 + (v & 3) * 2;
      af.u[v] = *(const unsigned*)&lA[(size_t)(wave * 16 + l16) * K + kk + ka];
    }
#pragma unroll
    for (int j = 0; j < 4; ++j) {
      int ng = nbase + j * 16 + l16;
      union { v16bf v; unsigned u[8]; } bfr;
#pragma unroll
      for (int v = 0; v < 8; ++v) {
        // B 16-bit 32x16 layout: same K order as A fragment
        int kb = (v >> 2) * 16 + half * 8 + (v & 3) * 2;
        bfr.u[v] = *(const unsigned*)(W + (size_t)ng * K + kk + kb);
      }
      acc[j] = __builtin_amdgcn_wmma_f32_16x16x32_bf16(false, af.v, false, bfr.v,
                                                       (short)0, acc[j], false, false);
    }
  }

#pragma unroll
  for (int j = 0; j < 4; ++j) {
#pragma unroll
    for (int r = 0; r < 8; ++r) {
      int m = blockM + wave * 16 + r + half * 8;  // C: VGPR r -> M = r + 8*(lane/16)
      int n = nbase + j * 16 + l16;
      if (planar) C[(size_t)n * ldc + m] = acc[j][r];
      else        C[(size_t)m * ldc + n] = acc[j][r];
    }
  }
}

// ---------------------------------------------------------------------------
// LayerNorm over 64 channels; one wave per token, lane owns channels c, c+32.
// ---------------------------------------------------------------------------
__global__ __launch_bounds__(256)
void ln1_kernel(const float* __restrict__ x, const float* __restrict__ w,
                const float* __restrict__ b, ushort_t* __restrict__ tokn) {
  int lane = threadIdx.x & 31, wave = threadIdx.x >> 5;
  int p  = blockIdx.x * 8 + wave;
  int bb = p / LSEQ, l = p % LSEQ;
  int c0 = lane, c1 = lane + 32;
  float v0 = x[(size_t)(bb * DIMC + c0) * LSEQ + l];
  float v1 = x[(size_t)(bb * DIMC + c1) * LSEQ + l];
  float s = v0 + v1, sq = v0 * v0 + v1 * v1;
  for (int off = 16; off; off >>= 1) { s += __shfl_xor(s, off, 32); sq += __shfl_xor(sq, off, 32); }
  float mu = s * (1.0f / DIMC);
  float rinv = rsqrtf(sq * (1.0f / DIMC) - mu * mu + EPSV);
  tokn[(size_t)p * DIMC + c0] = f2bf((v0 - mu) * rinv * w[c0] + b[c0]);
  tokn[(size_t)p * DIMC + c1] = f2bf((v1 - mu) * rinv * w[c1] + b[c1]);
}

// residual add + LN2: xres = x + mamba_out; t2n = LN(xres) as bf16 tokens
__global__ __launch_bounds__(256)
void res_ln2_kernel(const float* __restrict__ x, const float* __restrict__ tokout,
                    const float* __restrict__ w, const float* __restrict__ b,
                    float* __restrict__ xres, ushort_t* __restrict__ t2n) {
  int lane = threadIdx.x & 31, wave = threadIdx.x >> 5;
  int p  = blockIdx.x * 8 + wave;
  int bb = p / LSEQ, l = p % LSEQ;
  int c0 = lane, c1 = lane + 32;
  size_t i0 = (size_t)(bb * DIMC + c0) * LSEQ + l;
  size_t i1 = (size_t)(bb * DIMC + c1) * LSEQ + l;
  float v0 = x[i0] + tokout[(size_t)p * DIMC + c0];
  float v1 = x[i1] + tokout[(size_t)p * DIMC + c1];
  xres[i0] = v0; xres[i1] = v1;
  float s = v0 + v1, sq = v0 * v0 + v1 * v1;
  for (int off = 16; off; off >>= 1) { s += __shfl_xor(s, off, 32); sq += __shfl_xor(sq, off, 32); }
  float mu = s * (1.0f / DIMC);
  float rinv = rsqrtf(sq * (1.0f / DIMC) - mu * mu + EPSV);
  t2n[(size_t)p * DIMC + c0] = f2bf((v0 - mu) * rinv * w[c0] + b[c0]);
  t2n[(size_t)p * DIMC + c1] = f2bf((v1 - mu) * rinv * w[c1] + b[c1]);
}

// ---------------------------------------------------------------------------
// causal depthwise conv (window 4) over L + SiLU.  xz holds [xin | z] rows of 256.
// ---------------------------------------------------------------------------
__global__ void conv_silu_kernel(const float* __restrict__ xz, const float* __restrict__ cw,
                                 const float* __restrict__ cb, float* __restrict__ xcf,
                                 ushort_t* __restrict__ xcb) {
  int gid = blockIdx.x * blockDim.x + threadIdx.x;
  if (gid >= MTOK * DINNER) return;
  int d = gid & 127, p = gid >> 7;
  int bb = p / LSEQ, l = p % LSEQ;
  float acc = cb[d];
#pragma unroll
  for (int j = 0; j < 4; ++j) {
    int ll = l - 3 + j;
    if (ll >= 0) acc += cw[d * 4 + j] * xz[(size_t)(bb * LSEQ + ll) * 256 + d];
  }
  float sv = acc * sigm(acc);
  xcf[gid] = sv;
  xcb[gid] = f2bf(sv);
}

// dt = softplus(dt_r @ dt_proj_w^T + dt_proj_b);  dbc rows 64 wide (4|16|16|pad)
__global__ void dt_kernel(const float* __restrict__ dbc, const float* __restrict__ dtw,
                          const float* __restrict__ dtb, float* __restrict__ dt) {
  int gid = blockIdx.x * blockDim.x + threadIdx.x;
  if (gid >= MTOK * DINNER) return;
  int d = gid & 127, p = gid >> 7;
  float acc = dtb[d];
#pragma unroll
  for (int r = 0; r < 4; ++r) acc += dbc[(size_t)p * 64 + r] * dtw[d * 4 + r];
  dt[gid] = (acc > 20.f) ? acc : log1pf(__expf(acc));
}

// ---------------------------------------------------------------------------
// Selective scan: 256 (b,d) recurrences in parallel, 16 lanes = 16 states.
// ---------------------------------------------------------------------------
__global__ __launch_bounds__(256)
void scan_kernel(const float* __restrict__ xz, const float* __restrict__ xcf,
                 const float* __restrict__ dt, const float* __restrict__ dbc,
                 const float* __restrict__ A_log, const float* __restrict__ D_skip,
                 ushort_t* __restrict__ ybf) {
  int s    = threadIdx.x & 15;
  int pair = blockIdx.x * 16 + (threadIdx.x >> 4);
  int bb   = pair >> 7, d = pair & 127;
  float Aval = -__expf(A_log[d * 16 + s]);
  float Dv   = D_skip[d];
  float h = 0.f;
  for (int t = 0; t < LSEQ; ++t) {
    size_t base = (size_t)(bb * LSEQ + t);
    float xt  = xcf[base * 128 + d];
    float dtt = dt [base * 128 + d];
    float Bt  = dbc[base * 64 + 4 + s];
    float Ct  = dbc[base * 64 + 20 + s];
    h = __expf(dtt * Aval) * h + dtt * Bt * xt;
    float ys = h * Ct;
    ys += __shfl_xor(ys, 8, 16);
    ys += __shfl_xor(ys, 4, 16);
    ys += __shfl_xor(ys, 2, 16);
    ys += __shfl_xor(ys, 1, 16);
    if (s == 0) {
      float z = xz[base * 256 + 128 + d];
      ybf[base * 128 + d] = f2bf((ys + Dv * xt) * (z * sigm(z)));
    }
  }
}

// ---------------------------------------------------------------------------
// depthwise 3x3 on planar h (ch-major), exact-GELU gate, emit token-major bf16
// padded to K=192 (cols 170..191 zero) for the gdfn_out WMMA.
// ---------------------------------------------------------------------------
__global__ void dwgate_kernel(const float* __restrict__ h, const float* __restrict__ wdw,
                              ushort_t* __restrict__ gk) {
  int gid = blockIdx.x * blockDim.x + threadIdx.x;
  if (gid >= MTOK * 192) return;
  int c = gid % 192, p = gid / 192;
  if (c >= HIDN) { gk[gid] = 0; return; }
  int bb = p / LSEQ, pix = p % LSEQ;
  int y = pix >> 7, xcol = pix & 127;
  const float* w1 = wdw + (size_t)c * 9;
  const float* w2 = wdw + (size_t)(c + HIDN) * 9;
  const float* p1 = h + (size_t)c * MTOK + (size_t)bb * LSEQ;
  const float* p2 = h + (size_t)(c + HIDN) * MTOK + (size_t)bb * LSEQ;
  float h1 = 0.f, h2 = 0.f;
#pragma unroll
  for (int dy = 0; dy < 3; ++dy) {
    int yy = y + dy - 1;
    if (yy < 0 || yy >= HW) continue;
#pragma unroll
    for (int dx = 0; dx < 3; ++dx) {
      int xx = xcol + dx - 1;
      if (xx < 0 || xx >= HW) continue;
      int q = yy * HW + xx;
      h1 += w1[dy * 3 + dx] * p1[q];
      h2 += w2[dy * 3 + dx] * p2[q];
    }
  }
  float g = 0.5f * h1 * (1.0f + erff(h1 * 0.70710678f)) * h2;
  gk[gid] = f2bf(g);
}

// final residual:  out = xres + gdfn_out (token-major -> channel-major)
__global__ void final_kernel(const float* __restrict__ xres, const float* __restrict__ o,
                             float* __restrict__ out) {
  int gid = blockIdx.x * blockDim.x + threadIdx.x;
  if (gid >= BATCH * DIMC * LSEQ) return;
  int l = gid % LSEQ, c = (gid / LSEQ) % DIMC, bb = gid / (DIMC * LSEQ);
  out[gid] = xres[gid] + o[(size_t)(bb * LSEQ + l) * DIMC + c];
}

// ---------------------------------------------------------------------------
extern "C" void kernel_launch(void* const* d_in, const int* in_sizes, int n_in,
                              void* d_out, int out_size, void* d_ws, size_t ws_size,
                              hipStream_t stream) {
  const float* x         = (const float*)d_in[0];
  const float* ln1_w     = (const float*)d_in[1];
  const float* ln1_b     = (const float*)d_in[2];
  const float* in_proj_w = (const float*)d_in[3];
  const float* conv1d_w  = (const float*)d_in[4];
  const float* conv1d_b  = (const float*)d_in[5];
  const float* x_proj_w  = (const float*)d_in[6];
  const float* dt_proj_w = (const float*)d_in[7];
  const float* dt_proj_b = (const float*)d_in[8];
  const float* A_log     = (const float*)d_in[9];
  const float* D_skip    = (const float*)d_in[10];
  const float* out_proj_w= (const float*)d_in[11];
  const float* ln2_w     = (const float*)d_in[12];
  const float* ln2_b     = (const float*)d_in[13];
  const float* gdfn_in_w = (const float*)d_in[14];
  const float* gdfn_dw_w = (const float*)d_in[15];
  const float* gdfn_out_w= (const float*)d_in[16];

  char* ws = (char*)d_ws;
  const size_t MB = 1u << 20;
  ushort_t* tokn = (ushort_t*)(ws + 0);               // 4 MB (32768 x 64 bf16)
  ushort_t* Win  = (ushort_t*)(ws + 4*MB);            // 256x64
  ushort_t* Wxp  = (ushort_t*)(ws + 4*MB + 64*1024);  // 64x128 (pad 36->64)
  ushort_t* Wout = (ushort_t*)(ws + 4*MB + 128*1024); // 64x128
  ushort_t* Wg1  = (ushort_t*)(ws + 4*MB + 192*1024); // 384x64 (pad 340->384)
  ushort_t* Wg2  = (ushort_t*)(ws + 4*MB + 256*1024); // 64x192 (pad 170->192)
  float*    xz     = (float*)   (ws + 8*MB);    // 32 MB (32768 x 256)
  ushort_t* xcb    = (ushort_t*)(ws + 40*MB);   // 8 MB
  float*    xcf    = (float*)   (ws + 48*MB);   // 16 MB
  float*    dbc    = (float*)   (ws + 64*MB);   // 8 MB (32768 x 64)
  float*    dtb    = (float*)   (ws + 72*MB);   // 16 MB
  ushort_t* ybf    = (ushort_t*)(ws + 88*MB);   // 8 MB
  float*    tokout = (float*)   (ws + 96*MB);   // 8 MB
  float*    xres   = (float*)   (ws + 104*MB);  // 8 MB
  ushort_t* t2n    = (ushort_t*)(ws + 112*MB);  // 4 MB
  // post-scan reuse (xz/xcb/xcf/dbc dead):
  float*    hbuf = (float*)   (ws + 8*MB);      // 48 MB (384 planes x 32768)
  ushort_t* gk   = (ushort_t*)(ws + 56*MB);     // 12 MB (32768 x 192)
  float*    oout = (float*)   (ws + 88*MB);     // 8 MB (reuse ybf slot)

  const int T = 256;
  // weight prep (bf16, zero-padded to N multiple of 64)
  cvt_pad_kernel<<<(256*64 + T-1)/T, T, 0, stream>>>(in_proj_w, Win, 256, 64, 64, 256*64);
  cvt_pad_kernel<<<(64*128 + T-1)/T, T, 0, stream>>>(x_proj_w,  Wxp,  36, 128, 128, 64*128);
  cvt_pad_kernel<<<(64*128 + T-1)/T, T, 0, stream>>>(out_proj_w,Wout, 64, 128, 128, 64*128);
  cvt_pad_kernel<<<(384*64 + T-1)/T, T, 0, stream>>>(gdfn_in_w, Wg1, 340, 64, 64, 384*64);
  cvt_pad_kernel<<<(64*192 + T-1)/T, T, 0, stream>>>(gdfn_out_w,Wg2, 64, 170, 192, 64*192);

  // mamba path
  ln1_kernel<<<MTOK/8, T, 0, stream>>>(x, ln1_w, ln1_b, tokn);
  { dim3 g(MTOK/128, 4); wmma_gemm_bf16<<<g, T, 0, stream>>>(tokn, Win, xz, 64, 256, 0); }
  conv_silu_kernel<<<(MTOK*DINNER)/T, T, 0, stream>>>(xz, conv1d_w, conv1d_b, xcf, xcb);
  { dim3 g(MTOK/128, 1); wmma_gemm_bf16<<<g, T, 0, stream>>>(xcb, Wxp, dbc, 128, 64, 0); }
  dt_kernel<<<(MTOK*DINNER)/T, T, 0, stream>>>(dbc, dt_proj_w, dt_proj_b, dtb);
  scan_kernel<<<16, T, 0, stream>>>(xz, xcf, dtb, dbc, A_log, D_skip, ybf);
  { dim3 g(MTOK/128, 1); wmma_gemm_bf16<<<g, T, 0, stream>>>(ybf, Wout, tokout, 128, 64, 0); }
  res_ln2_kernel<<<MTOK/8, T, 0, stream>>>(x, tokout, ln2_w, ln2_b, xres, t2n);

  // gdfn path
  { dim3 g(MTOK/128, 6); wmma_gemm_bf16<<<g, T, 0, stream>>>(t2n, Wg1, hbuf, 64, MTOK, 1); }
  dwgate_kernel<<<(MTOK*192)/T, T, 0, stream>>>(hbuf, gdfn_dw_w, gk);
  { dim3 g(MTOK/128, 1); wmma_gemm_bf16<<<g, T, 0, stream>>>(gk, Wg2, oout, 192, 64, 0); }
  final_kernel<<<(BATCH*DIMC*LSEQ)/T, T, 0, stream>>>(xres, oout, (float*)d_out);
}
